// GRUDynamics_90812788506735
// MI455X (gfx1250) — compile-verified, matrix operand order checked by reading, not run on hardware
//
#include <hip/hip_runtime.h>
#include <cstdint>
#include <cstddef>

typedef __bf16 v16bf __attribute__((ext_vector_type(16)));
typedef float  v8f   __attribute__((ext_vector_type(8)));

#define B_TOTAL 2048
#define LATENT  256
#define HIDDEN  512
#define G3      1536
#define BT      32            // batch rows per workgroup
#define NWG     (B_TOTAL/BT)  // 64
#define NTHREADS 256          // 8 wave32
#define HSTR    528           // 512 + 16 pad (keeps 32B alignment, staggers banks)
#define PSTR    272           // 256 + 16 pad
#define NT_G    96            // N-tiles across 3H=1536
#define KT_P    8             // K-tiles over LATENT (256/32)
#define KT_H    16            // K-tiles over HIDDEN (512/32)
#define NT_P    16            // N-tiles over LATENT (256/16)

__device__ __forceinline__ unsigned short f2bf(float f) {
  union { float f; unsigned u; } v; v.f = f;
  unsigned r = v.u + 0x7FFFu + ((v.u >> 16) & 1u);
  return (unsigned short)(r >> 16);
}
__device__ __forceinline__ float bf2f(unsigned short h) {
  union { unsigned u; float f; } v; v.u = ((unsigned)h) << 16;
  return v.f;
}
__device__ __forceinline__ float sigm_(float x) { return 1.f / (1.f + __expf(-x)); }
__device__ __forceinline__ float tanh_(float x) {
  float e = __expf(2.f * x);
  return (e - 1.f) / (e + 1.f);
}
__device__ __forceinline__ v8f zero8() {
  v8f r = {0.f,0.f,0.f,0.f,0.f,0.f,0.f,0.f}; return r;
}
__device__ __forceinline__ v8f wmma_bf16(v16bf a, v16bf b, v8f c) {
  // D = A(16x32 bf16) x B(32x16 bf16) + C(16x16 f32)
  return __builtin_amdgcn_wmma_f32_16x16x32_bf16(false, a, false, b, (short)0, c, false, false);
}

// Pack W (N_total x K row-major, fp32) into WMMA-B tiles of 32x16 bf16.
// Tile (kc, nt) contiguous 512 bf16; lane l holds 16 contiguous values.
__global__ void pack_b_tiles(const float* __restrict__ W, unsigned short* __restrict__ dst,
                             int Ktiles, int Ntiles, int src_stride) {
  int total = Ktiles * Ntiles * 512;
  for (int i = blockIdx.x * blockDim.x + threadIdx.x; i < total; i += gridDim.x * blockDim.x) {
    int e  = i & 15;
    int l  = (i >> 4) & 31;
    int t  = i >> 9;
    int nt = t % Ntiles;
    int kc = t / Ntiles;
    int row = nt * 16 + (l & 15);
    int col = kc * 32 + ((l >> 4) << 4) + e;
    dst[i] = f2bf(W[(size_t)row * src_stride + col]);
  }
}

// lat/lon columns of W_ih (step-invariant) + pre-combined gate biases.
__global__ void prep_bias(const float* __restrict__ W_ih, const float* __restrict__ b_ih,
                          const float* __restrict__ b_hh,
                          float* __restrict__ wlat, float* __restrict__ wlon,
                          float* __restrict__ bsr, float* __restrict__ bsz,
                          float* __restrict__ bin, float* __restrict__ bhn) {
  int n = blockIdx.x * blockDim.x + threadIdx.x;
  if (n < G3) {
    wlat[n] = W_ih[(size_t)n * 258 + 256];
    wlon[n] = W_ih[(size_t)n * 258 + 257];
  }
  if (n < HIDDEN) {
    bsr[n] = b_ih[n] + b_hh[n];
    bsz[n] = b_ih[HIDDEN + n] + b_hh[HIDDEN + n];
    bin[n] = b_ih[2 * HIDDEN + n];
    bhn[n] = b_hh[2 * HIDDEN + n];
  }
}

__global__ __launch_bounds__(NTHREADS)
void gru_persistent(const float* __restrict__ p0, const float* __restrict__ lat,
                    const float* __restrict__ lon, const float* __restrict__ b_proj,
                    const unsigned short* __restrict__ Wp,  // packed W_ih[:, :256], 8x96 tiles
                    const unsigned short* __restrict__ Whh, // packed W_hh, 16x96 tiles
                    const unsigned short* __restrict__ Wpr, // packed W_proj, 16x16 tiles
                    const float* __restrict__ wlat, const float* __restrict__ wlon,
                    const float* __restrict__ bsr, const float* __restrict__ bsz,
                    const float* __restrict__ bin, const float* __restrict__ bhn,
                    const int* __restrict__ n_steps_p, float* __restrict__ out) {
  __shared__ __attribute__((aligned(32))) unsigned short hA[2][BT * HSTR]; // h, bf16, dbl-buffered
  __shared__ __attribute__((aligned(32))) unsigned short pA[BT * PSTR];    // p, bf16
  __shared__ float s_lat[BT], s_lon[BT];

  const int tid   = threadIdx.x;
  const int lane  = tid & 31;
  const int w     = tid >> 5;      // wave 0..7, owns hidden cols [w*64, w*64+64)
  const int laneM = lane & 15;
  const int laneH = lane >> 4;     // 0/1 (K half within A tile)
  const int row0  = blockIdx.x * BT;

  // init: out[0] = p0 slice, pack p0 -> bf16 LDS, h0 = 0, load lat/lon
  for (int i = tid; i < BT * LATENT; i += NTHREADS) {
    int m = i >> 8, c = i & 255;
    float pv = p0[(size_t)(row0 + m) * LATENT + c];
    out[(size_t)(row0 + m) * LATENT + c] = pv;
    pA[m * PSTR + c] = f2bf(pv);
  }
  for (int i = tid; i < BT * HSTR; i += NTHREADS) hA[0][i] = 0;
  if (tid < BT) { s_lat[tid] = lat[row0 + tid]; s_lon[tid] = lon[row0 + tid]; }
  __syncthreads();

  // step-invariant per-lane row conditioning (kills 128 ds_loads/step)
  float laR[2][8], loR[2][8];
  #pragma unroll
  for (int mt = 0; mt < 2; ++mt)
    #pragma unroll
    for (int v = 0; v < 8; ++v) {
      const int m = mt * 16 + laneH * 8 + v;
      laR[mt][v] = s_lat[m];
      loR[mt][v] = s_lon[m];
    }

  // hoisted operand base pointers -> all hot-loop loads use immediate offsets
  const unsigned short* pA0 = &pA[laneM * PSTR + laneH * 16];

  const int nsteps = *n_steps_p;
  int buf = 0;
  for (int s = 0; s < nsteps; ++s) {
    const unsigned short* hc = hA[buf];
    unsigned short*       hn = hA[buf ^ 1];
    const unsigned short* hc0 = &hc[laneM * HSTR + laneH * 16];

    // ---- gate phase: 4 hidden sub-tiles of 16 cols per wave ----
    for (int t = 0; t < 4; ++t) {
      const int nth = w * 4 + t;   // hidden N-tile 0..31
      const int jj  = nth * 16 + laneM;
      const unsigned short* wpB  = Wp  + (size_t)nth * 512 + (size_t)lane * 16;
      const unsigned short* whhB = Whh + (size_t)nth * 512 + (size_t)lane * 16;

      v8f ar[2]  = {zero8(), zero8()};  // r gate (gi+gh fused)
      v8f az[2]  = {zero8(), zero8()};  // z gate (gi+gh fused)
      v8f ani[2] = {zero8(), zero8()};  // n gate, input part
      v8f anh[2] = {zero8(), zero8()};  // n gate, hidden part (kept separate for r*gh_n)

      #pragma unroll
      for (int kc = 0; kc < KT_P; ++kc) {   // p @ W_ih[:, :256]^T
        v16bf a0 = *(const v16bf*)(pA0 + kc * 32);
        v16bf a1 = *(const v16bf*)(pA0 + 16 * PSTR + kc * 32);
        v16bf br = *(const v16bf*)(wpB + kc * (NT_G * 512));             // gate r
        v16bf bz = *(const v16bf*)(wpB + kc * (NT_G * 512) + 32 * 512);  // gate z
        v16bf bn = *(const v16bf*)(wpB + kc * (NT_G * 512) + 64 * 512);  // gate n
        ar[0]  = wmma_bf16(a0, br, ar[0]);   ar[1]  = wmma_bf16(a1, br, ar[1]);
        az[0]  = wmma_bf16(a0, bz, az[0]);   az[1]  = wmma_bf16(a1, bz, az[1]);
        ani[0] = wmma_bf16(a0, bn, ani[0]);  ani[1] = wmma_bf16(a1, bn, ani[1]);
      }
      #pragma unroll
      for (int kc = 0; kc < KT_H; ++kc) {   // h @ W_hh^T
        v16bf a0 = *(const v16bf*)(hc0 + kc * 32);
        v16bf a1 = *(const v16bf*)(hc0 + 16 * HSTR + kc * 32);
        if (kc < KT_H - 1)
          __builtin_prefetch(whhB + (kc + 1) * (NT_G * 512), 0, 0);      // next K-tile
        v16bf br = *(const v16bf*)(whhB + kc * (NT_G * 512));
        v16bf bz = *(const v16bf*)(whhB + kc * (NT_G * 512) + 32 * 512);
        v16bf bn = *(const v16bf*)(whhB + kc * (NT_G * 512) + 64 * 512);
        ar[0]  = wmma_bf16(a0, br, ar[0]);   ar[1]  = wmma_bf16(a1, br, ar[1]);
        az[0]  = wmma_bf16(a0, bz, az[0]);   az[1]  = wmma_bf16(a1, bz, az[1]);
        anh[0] = wmma_bf16(a0, bn, anh[0]);  anh[1] = wmma_bf16(a1, bn, anh[1]);
      }

      // per-column scalars (L2/L1 resident)
      const float cr = bsr[jj], cz = bsz[jj], cin = bin[jj], chn = bhn[jj];
      const float wlr = wlat[jj], wlz = wlat[512 + jj], wln = wlat[1024 + jj];
      const float wor = wlon[jj], woz = wlon[512 + jj], won = wlon[1024 + jj];

      // elementwise GRU update; C-layout: vgpr v, lanes 0-15 -> M=v, lanes 16-31 -> M=v+8
      const int eb = laneH * 8 * HSTR + jj;   // immediate-foldable LDS addressing
      #pragma unroll
      for (int mt = 0; mt < 2; ++mt) {
        #pragma unroll
        for (int v = 0; v < 8; ++v) {
          const float la = laR[mt][v], lo = loR[mt][v];
          float r  = sigm_(ar[mt][v] + cr + la * wlr + lo * wor);
          float z  = sigm_(az[mt][v] + cz + la * wlz + lo * woz);
          float gh = anh[mt][v] + chn;
          float nn = tanh_(ani[mt][v] + cin + la * wln + lo * won + r * gh);
          float ho = bf2f(hc[eb + (mt * 16 + v) * HSTR]);
          float hv = (1.f - z) * nn + z * ho;
          hn[eb + (mt * 16 + v) * HSTR] = f2bf(hv);
        }
      }
    }
    __syncthreads();   // all of h_new visible before projection

    // ---- projection: p_new = h_new @ W_proj^T + b_proj; wave w owns cols [w*32, w*32+32)
    const unsigned short* hn0 = &hn[laneM * HSTR + laneH * 16];
    #pragma unroll
    for (int ntl = 0; ntl < 2; ++ntl) {
      const int np = w * 2 + ntl;   // latent N-tile 0..15
      const unsigned short* wprB = Wpr + (size_t)np * 512 + (size_t)lane * 16;
      v8f ac[2] = {zero8(), zero8()};
      #pragma unroll
      for (int kc = 0; kc < KT_H; ++kc) {
        v16bf a0 = *(const v16bf*)(hn0 + kc * 32);
        v16bf a1 = *(const v16bf*)(hn0 + 16 * HSTR + kc * 32);
        v16bf bp = *(const v16bf*)(wprB + kc * (NT_P * 512));
        ac[0] = wmma_bf16(a0, bp, ac[0]);
        ac[1] = wmma_bf16(a1, bp, ac[1]);
      }
      const int ncol = np * 16 + laneM;
      const float bpv = b_proj[ncol];
      float* op = out + ((size_t)(s + 1) * B_TOTAL + row0 + laneH * 8) * LATENT + ncol;
      const int pb = laneH * 8 * PSTR + ncol;
      #pragma unroll
      for (int mt = 0; mt < 2; ++mt) {
        #pragma unroll
        for (int v = 0; v < 8; ++v) {
          float pv = ac[mt][v] + bpv;
          op[(size_t)(mt * 16 + v) * LATENT] = pv;
          pA[pb + (mt * 16 + v) * PSTR] = f2bf(pv);
        }
      }
    }
    __syncthreads();   // pA complete before next step's gate phase
    buf ^= 1;
  }
}

extern "C" void kernel_launch(void* const* d_in, const int* in_sizes, int n_in,
                              void* d_out, int out_size, void* d_ws, size_t ws_size,
                              hipStream_t stream) {
  const float* p0     = (const float*)d_in[0];
  const float* lat    = (const float*)d_in[1];
  const float* lon    = (const float*)d_in[2];
  const float* W_ih   = (const float*)d_in[3];
  const float* W_hh   = (const float*)d_in[4];
  const float* b_ih   = (const float*)d_in[5];
  const float* b_hh   = (const float*)d_in[6];
  const float* W_proj = (const float*)d_in[7];
  const float* b_proj = (const float*)d_in[8];
  const int*   nsteps = (const int*)d_in[9];
  float* out = (float*)d_out;

  // workspace layout (bf16-packed weights + fused biases, ~2.65 MB; L2-resident)
  unsigned short* Wp  = (unsigned short*)d_ws;            // 8*96*512  bf16
  unsigned short* Whh = Wp  + (size_t)KT_P * NT_G * 512;  // 16*96*512 bf16
  unsigned short* Wpr = Whh + (size_t)KT_H * NT_G * 512;  // 16*16*512 bf16
  float* wlat = (float*)(Wpr + (size_t)KT_H * NT_P * 512);
  float* wlon = wlat + G3;
  float* bsr  = wlon + G3;
  float* bsz  = bsr + HIDDEN;
  float* bin  = bsz + HIDDEN;
  float* bhn  = bin + HIDDEN;

  pack_b_tiles<<<512, 256, 0, stream>>>(W_ih,   Wp,  KT_P, NT_G, 258);
  pack_b_tiles<<<512, 256, 0, stream>>>(W_hh,   Whh, KT_H, NT_G, 512);
  pack_b_tiles<<<64,  256, 0, stream>>>(W_proj, Wpr, KT_H, NT_P, 512);
  prep_bias<<<(G3 + 255) / 256, 256, 0, stream>>>(W_ih, b_ih, b_hh,
                                                  wlat, wlon, bsr, bsz, bin, bhn);

  gru_persistent<<<NWG, NTHREADS, 0, stream>>>(p0, lat, lon, b_proj,
                                               Wp, Whh, Wpr, wlat, wlon,
                                               bsr, bsz, bin, bhn, nsteps, out);
}
